// CustomQLoRABigNet_10900626998009
// MI455X (gfx1250) — matrix operand order repaired and checked by reading, not compiled, not run on hardware
//
#include <hip/hip_runtime.h>
#include <hip/hip_bf16.h>

typedef _Float16 half8  __attribute__((ext_vector_type(8)));
typedef _Float16 half16 __attribute__((ext_vector_type(16)));
typedef float    float8 __attribute__((ext_vector_type(8)));
typedef float    floatx4 __attribute__((ext_vector_type(4)));

#define DIM   1024
#define NLIN  18
#define RANK  32
#define MT    32            // rows per workgroup
#define HST   (DIM + 8)     // padded LDS row stride (halves) -> conflict-free ds_load_b128
#define TST   40            // t-buffer row stride (halves)
#define SMEM_BYTES ((3 * MT * HST + MT * TST) * 2 + 512 * 4)   // 202,752 B

// ---------------------------------------------------------------------------
// WMMA helpers
// ---------------------------------------------------------------------------
__device__ __forceinline__ float8 wmma16(half16 a, half16 b, float8 c) {
  return __builtin_amdgcn_wmma_f32_16x16x32_f16(false, a, false, b, (short)0, c,
                                                false, false);
}

// A-fragment (16x32 f16): lanes 0-15 -> M=lane, K in {0..15};
// lanes 16-31 -> M=lane-16, K in {16..31}. Source is row-major [M][stride].
__device__ __forceinline__ half16 load_afrag(const _Float16* base, int strideHalves,
                                             int mtile, int kc, int lane) {
  int m = (lane & 15) + mtile * 16;
  int kh = kc * 32 + ((lane & 16) ? 16 : 0);
  const _Float16* p = base + m * strideHalves + kh;
  half8 lo = *(const half8*)(p);
  half8 hi = *(const half8*)(p + 8);
  return __builtin_shufflevector(lo, hi, 0, 1, 2, 3, 4, 5, 6, 7, 8, 9, 10, 11, 12,
                                 13, 14, 15);
}

// ---------------------------------------------------------------------------
// Prepass 1: dequant int4-in-int32 weights -> f16, pre-swizzled B fragments.
// Fragment for output-tile nt (16 cols), K-chunk kc (32):
//   lane = nloc (+16 if k%32>=16), halves i=0..15 hold K = (k%16)=i.
// One thread handles one scale group (16 consecutive K) => one 32B store.
// ---------------------------------------------------------------------------
__global__ void k_dequant_pack_w(const int* __restrict__ wq,
                                 const float* __restrict__ scales,
                                 _Float16* __restrict__ pW) {
  int g = blockIdx.x * blockDim.x + threadIdx.x;
  if (g >= NLIN * 1024 * 64) return;
  int li = g / (1024 * 64);
  int rem = g - li * (1024 * 64);
  int n = rem >> 6;        // output row of W
  int gk = rem & 63;       // 16-wide K group
  float s = scales[(size_t)li * 65536 + n * 64 + gk];
  const int* q = wq + (((size_t)li * 1024 + n) * 1024 + gk * 16);
  half16 v;
#pragma unroll
  for (int t = 0; t < 16; t++) v[t] = (_Float16)((float)(q[t] - 8) * s);
  int nt = n >> 4, nloc = n & 15;
  int kc = gk >> 1;
  int lane = nloc + ((gk & 1) ? 16 : 0);
  *(half16*)(pW + ((size_t)((li * 64 + nt) * 32 + kc)) * 512 + lane * 16) = v;
}

// Prepass 2: pack lora_a [NLIN][RANK][DIM] as B for t = h @ la^T  (B[k][r]=la[r][k])
__global__ void k_pack_la(const float* __restrict__ la, _Float16* __restrict__ pLA) {
  int g = blockIdx.x * blockDim.x + threadIdx.x;
  if (g >= NLIN * RANK * 64) return;
  int li = g / (RANK * 64);
  int rem = g - li * (RANK * 64);
  int r = rem >> 6;
  int gk = rem & 63;
  const float* src = la + (((size_t)li * RANK + r) * 1024 + gk * 16);
  half16 v;
#pragma unroll
  for (int t = 0; t < 16; t++) v[t] = (_Float16)src[t];
  int nr = r >> 4, rloc = r & 15;
  int kc = gk >> 1;
  int lane = rloc + ((gk & 1) ? 16 : 0);
  *(half16*)(pLA + ((size_t)((li * 2 + nr) * 32 + kc)) * 512 + lane * 16) = v;
}

// Prepass 3: pack lora_b [NLIN][DIM][RANK] as B for out += t @ lb^T (B[r][n]=lb[n][r])
__global__ void k_pack_lb(const float* __restrict__ lb, _Float16* __restrict__ pLB) {
  int g = blockIdx.x * blockDim.x + threadIdx.x;
  if (g >= NLIN * 1024) return;
  int li = g >> 10;
  int n = g & 1023;
  const float* src = lb + (((size_t)li * 1024 + n) * RANK);
  int nt = n >> 4, nloc = n & 15;
  _Float16* dst = pLB + ((size_t)(li * 64 + nt)) * 512;
  half16 v0, v1;
#pragma unroll
  for (int t = 0; t < 16; t++) {
    v0[t] = (_Float16)src[t];
    v1[t] = (_Float16)src[16 + t];
  }
  *(half16*)(dst + nloc * 16) = v0;          // K = 0..15   -> lanes 0-15
  *(half16*)(dst + (nloc + 16) * 16) = v1;   // K = 16..31  -> lanes 16-31
}

// ---------------------------------------------------------------------------
// Main fused kernel: each workgroup owns 32 batch rows and chains all 18
// layers (+LoRA, bias, ReLU, residual, LayerNorm) in LDS. Weights stream from
// L2-resident pre-packed f16 fragments.
// ---------------------------------------------------------------------------
__global__ __launch_bounds__(256, 1) void k_qlora_chain(
    const float* __restrict__ x, const float* __restrict__ bias,
    const float* __restrict__ gamma, const float* __restrict__ beta,
    const _Float16* __restrict__ pW, const _Float16* __restrict__ pLA,
    const _Float16* __restrict__ pLB, float* __restrict__ out) {
  extern __shared__ char smem[];
  _Float16* buf0 = (_Float16*)smem;            // [MT][HST]
  _Float16* buf1 = buf0 + MT * HST;            // [MT][HST]
  _Float16* rbuf = buf1 + MT * HST;            // residual [MT][HST]
  _Float16* tbuf = rbuf + MT * HST;            // lora mid [MT][TST]
  float* red = (float*)(tbuf + MT * TST);      // 512 floats

  const int tid = threadIdx.x;
  const int lane = tid & 31;
  const int wave = tid >> 5;
  const int row = tid >> 3;  // 0..31
  const int seg = tid & 7;   // 0..7 : 128-col segment
  const size_t growbase = (size_t)blockIdx.x * MT;

  // ---- load x slab -> buf0 (f16) ----
  {
    const floatx4* xp = (const floatx4*)(x + (growbase + row) * DIM + seg * 128);
    _Float16* hp = buf0 + row * HST + seg * 128;
#pragma unroll 8
    for (int c = 0; c < 32; c++) {
      floatx4 v = __builtin_nontemporal_load(xp + c);
      hp[c * 4 + 0] = (_Float16)v[0];
      hp[c * 4 + 1] = (_Float16)v[1];
      hp[c * 4 + 2] = (_Float16)v[2];
      hp[c * 4 + 3] = (_Float16)v[3];
    }
  }
  __syncthreads();

  _Float16* cur = buf0;
  _Float16* nxt = buf1;

  for (int blk = 0; blk < 6; blk++) {
    // ---- residual copy r = h ----
    {
      const half8* s = (const half8*)(cur + row * HST + seg * 128);
      half8* d = (half8*)(rbuf + row * HST + seg * 128);
#pragma unroll
      for (int c = 0; c < 16; c++) d[c] = s[c];
    }
    __syncthreads();

    for (int lj = 0; lj < 3; lj++) {
      const int li = blk * 3 + lj;

      // ---- Phase A: t = h @ la^T  (32x32, waves 0..3, one tile each) ----
      if (wave < 4) {
        const int mt = wave >> 1, nr = wave & 1;
        float8 acc = {0.f, 0.f, 0.f, 0.f, 0.f, 0.f, 0.f, 0.f};
        const _Float16* pla = pLA + ((size_t)(li * 2 + nr)) * 32 * 512 + lane * 16;
        for (int kc = 0; kc < 32; kc++) {
          half16 a = load_afrag(cur, HST, mt, kc, lane);
          half16 b = *(const half16*)(pla + (size_t)kc * 512);
          acc = wmma16(a, b, acc);
        }
        const int nloc = lane & 15;
        const int ro = (lane & 16) ? 8 : 0;
#pragma unroll
        for (int j = 0; j < 8; j++)
          tbuf[(mt * 16 + j + ro) * TST + nr * 16 + nloc] = (_Float16)acc[j];
      }
      __syncthreads();

      // ---- Phase B: out = h @ W^T + bias + t @ lb^T (+relu / +residual) ----
      for (int iter = 0; iter < 4; iter++) {
        const int nt0 = (iter * 256 + wave * 32) >> 4;  // first 16-col tile
        float8 acc[2][2];
        // seed accumulators with LoRA (single K=32 WMMA per tile)
        {
          half16 at0 = load_afrag(tbuf, TST, 0, 0, lane);
          half16 at1 = load_afrag(tbuf, TST, 1, 0, lane);
          const float8 z = {0.f, 0.f, 0.f, 0.f, 0.f, 0.f, 0.f, 0.f};
#pragma unroll
          for (int nn = 0; nn < 2; nn++) {
            half16 bl = *(const half16*)(pLB + ((size_t)(li * 64 + nt0 + nn)) * 512 +
                                         lane * 16);
            acc[0][nn] = wmma16(at0, bl, z);
            acc[1][nn] = wmma16(at1, bl, z);
          }
        }
        // main K=1024 reduction against dequantized weights (from L2)
        const _Float16* pw0 =
            pW + ((size_t)((li * 64 + nt0) * 32)) * 512 + lane * 16;
        for (int kc = 0; kc < 32; kc++) {
          half16 a0 = load_afrag(cur, HST, 0, kc, lane);
          half16 a1 = load_afrag(cur, HST, 1, kc, lane);
#pragma unroll
          for (int nn = 0; nn < 2; nn++) {
            half16 b =
                *(const half16*)(pw0 + ((size_t)nn * 32 + kc) * 512);
            acc[0][nn] = wmma16(a0, b, acc[0][nn]);
            acc[1][nn] = wmma16(a1, b, acc[1][nn]);
          }
        }
        // epilogue: bias (+relu or +residual), write f16 to ping-pong buffer
        const int nloc = lane & 15;
        const int ro = (lane & 16) ? 8 : 0;
#pragma unroll
        for (int mt2 = 0; mt2 < 2; mt2++) {
#pragma unroll
          for (int nn = 0; nn < 2; nn++) {
            const int n = (nt0 + nn) * 16 + nloc;
            const float bv = bias[li * DIM + n];
#pragma unroll
            for (int j = 0; j < 8; j++) {
              const int m = mt2 * 16 + j + ro;
              float v = acc[mt2][nn][j] + bv;
              if (lj < 2)
                v = fmaxf(v, 0.f);
              else
                v += (float)rbuf[m * HST + n];  // h + r on last linear
              nxt[m * HST + n] = (_Float16)v;
            }
          }
        }
      }
      __syncthreads();
      _Float16* tmp = cur;
      cur = nxt;
      nxt = tmp;
    }

    // ---- LayerNorm (blocks 0..4), f32 math, in place ----
    if (blk < 5) {
      _Float16* hp = cur + row * HST + seg * 128;
      float sum = 0.f, ss = 0.f;
      for (int c = 0; c < 128; c++) {
        float v = (float)hp[c];
        sum += v;
        ss += v * v;
      }
      red[row * 8 + seg] = sum;
      red[256 + row * 8 + seg] = ss;
      __syncthreads();
      if (seg == 0) {
        float s = 0.f, q = 0.f;
#pragma unroll
        for (int k2 = 0; k2 < 8; k2++) {
          s += red[row * 8 + k2];
          q += red[256 + row * 8 + k2];
        }
        float mean = s * (1.f / 1024.f);
        float var = q * (1.f / 1024.f) - mean * mean;
        red[row * 8] = mean;
        red[256 + row * 8] = rsqrtf(var + 1e-5f);
      }
      __syncthreads();
      const float mean = red[row * 8];
      const float rstd = red[256 + row * 8];
      for (int c = 0; c < 128; c++) {
        const int n = seg * 128 + c;
        float v = ((float)hp[c] - mean) * rstd * gamma[blk * DIM + n] +
                  beta[blk * DIM + n];
        hp[c] = (_Float16)v;
      }
      __syncthreads();
    }
  }

  // ---- write result slab (f32) ----
  {
    floatx4* op = (floatx4*)(out + (growbase + row) * DIM + seg * 128);
    const _Float16* hp = cur + row * HST + seg * 128;
#pragma unroll 8
    for (int c = 0; c < 32; c++) {
      floatx4 v;
      v[0] = (float)hp[c * 4 + 0];
      v[1] = (float)hp[c * 4 + 1];
      v[2] = (float)hp[c * 4 + 2];
      v[3] = (float)hp[c * 4 + 3];
      __builtin_nontemporal_store(v, op + c);
    }
  }
}

// ---------------------------------------------------------------------------
extern "C" void kernel_launch(void* const* d_in, const int* in_sizes, int n_in,
                              void* d_out, int out_size, void* d_ws, size_t ws_size,
                              hipStream_t stream) {
  const float* x = (const float*)d_in[0];
  const int* wq = (const int*)d_in[1];
  const float* scales = (const float*)d_in[2];
  const float* bias = (const float*)d_in[3];
  const float* lora_a = (const float*)d_in[4];
  const float* lora_b = (const float*)d_in[5];
  const float* gamma = (const float*)d_in[6];
  const float* beta = (const float*)d_in[7];
  float* out = (float*)d_out;

  // workspace layout (halves)
  _Float16* pW = (_Float16*)d_ws;                        // 18*64*32*512 = 18,874,368 h
  _Float16* pLA = pW + (size_t)NLIN * 64 * 32 * 512;     // 18*2*32*512  =    589,824 h
  _Float16* pLB = pLA + (size_t)NLIN * 2 * 32 * 512;     // 18*64*512    =    589,824 h

  (void)hipFuncSetAttribute((const void*)k_qlora_chain,
                            hipFuncAttributeMaxDynamicSharedMemorySize, SMEM_BYTES);

  k_dequant_pack_w<<<(NLIN * 1024 * 64 + 255) / 256, 256, 0, stream>>>(wq, scales, pW);
  k_pack_la<<<(NLIN * RANK * 64 + 255) / 256, 256, 0, stream>>>(lora_a, pLA);
  k_pack_lb<<<(NLIN * 1024 + 255) / 256, 256, 0, stream>>>(lora_b, pLB);

  k_qlora_chain<<<32768 / MT, 256, SMEM_BYTES, stream>>>(x, bias, gamma, beta, pW,
                                                         pLA, pLB, out);
}
